// MultiHeadAttention_36558761624388
// MI455X (gfx1250) — matrix operand order repaired and checked by reading, not compiled, not run on hardware
//
#include <hip/hip_runtime.h>
#include <cstdint>
#include <cstddef>

typedef __attribute__((ext_vector_type(16))) __bf16 v16bf;
typedef __attribute__((ext_vector_type(8)))  __bf16 v8bf;
typedef __attribute__((ext_vector_type(8)))  float  v8f;
typedef __attribute__((ext_vector_type(4)))  unsigned int v4u;
typedef __attribute__((ext_vector_type(8)))  int    v8i;
typedef __attribute__((ext_vector_type(4)))  int    v4i;

#define HIDDEN 2048
#define NHEADS 16
#define HDIM   128
#define BATCH  2
#define SEQ    2048
#define MROWS  (BATCH * SEQ)   // 4096
#define KT     64              // K-slab per TDM stage

// ---------------------------------------------------------------------------
// helpers
// ---------------------------------------------------------------------------
__device__ __forceinline__ unsigned short f2bf(float f) {
  unsigned u = __builtin_bit_cast(unsigned, f);
  u += 0x7FFFu + ((u >> 16) & 1u);       // round-to-nearest-even
  return (unsigned short)(u >> 16);
}

__device__ __forceinline__ v8f wmma_bf16(v16bf a, v16bf b, v8f c) {
  return __builtin_amdgcn_wmma_f32_16x16x32_bf16(
      /*neg_a=*/false, a, /*neg_b=*/false, b,
      /*c_mod=*/(short)0, c, /*reuse_a=*/false, /*reuse_b=*/false);
}

// A-operand 16x32 fragment from row-major 16-bit source (ISA 7.12.2):
// lanes 0-15: row=lane, K={0..7,16..23}; lanes 16-31: row=lane-16, K={8..15,24..31}
__device__ __forceinline__ v16bf load_fragA(const unsigned short* base, int stride) {
  const int lane = threadIdx.x & 31;
  const int r    = lane & 15;
  const int kofs = (lane >> 4) * 8;
  const unsigned short* p = base + (size_t)r * stride + kofs;
  union { v16bf v; v8bf h[2]; } u;
  u.h[0] = *(const v8bf*)(p);
  u.h[1] = *(const v8bf*)(p + 16);
  return u.v;
}

// B-operand 32x16 (KxN): lane n holds K of source row n in the same order as A:
// lanes 0-15: K={0..7,16..23}; lanes 16-31: K={8..15,24..31}
__device__ __forceinline__ v16bf load_fragB(const unsigned short* base, int stride) {
  const int lane = threadIdx.x & 31;
  const int n    = lane & 15;
  const int kofs = (lane >> 4) * 8;
  const unsigned short* p = base + (size_t)n * stride + kofs;
  union { v16bf v; v8bf h[2]; } u;
  u.h[0] = *(const v8bf*)(p);
  u.h[1] = *(const v8bf*)(p + 16);
  return u.v;
}

// ---------------------------------------------------------------------------
// Tensor Data Mover: 2D tile load Global -> LDS (D# per ISA cdna5 ch.8)
//   tile = tile_d1 rows x tile_d0 elems, 2-byte elements, row-major source with
//   row stride `stride_elems`; packed row-major into LDS at lds_off.
// ---------------------------------------------------------------------------
__device__ __forceinline__ void tdm_load_2d(unsigned lds_off, const void* gaddr,
                                            unsigned tile_d0, unsigned tile_d1,
                                            unsigned stride_elems) {
  const unsigned long long ga = (unsigned long long)gaddr;
  v4u g0;
  g0[0] = 1u;                                     // count=1 (valid), user mode
  g0[1] = lds_off;                                // lds_addr (bytes)
  g0[2] = (unsigned)(ga & 0xFFFFFFFFu);           // global_addr[31:0]
  g0[3] = (unsigned)((ga >> 32) & 0x01FFFFFFu)    // global_addr[56:32]
          | (2u << 30);                           // type=2 ("image")
  v8i g1;
  g1[0] = (int)(1u << 16);                        // wg_mask=0, data_size=1 (2B)
  g1[1] = (int)((stride_elems & 0xFFFFu) << 16);  // tensor_dim0[15:0]
  g1[2] = (int)(((stride_elems >> 16) & 0xFFFFu)  // tensor_dim0[31:16]
          | 0x80000000u);                         // tensor_dim1[15:0] = 32768
  g1[3] = (int)((tile_d0 & 0xFFFFu) << 16);       // tensor_dim1[31:16]=0, tile_dim0
  g1[4] = (int)(tile_d1 & 0xFFFFu);               // tile_dim1, tile_dim2=0
  g1[5] = (int)stride_elems;                      // tensor_dim0_stride[31:0]
  g1[6] = 0;                                      // dim0_stride[47:32], dim1_stride lo
  g1[7] = 0;
  v4i gz = (v4i)0;
#if __has_include(<hip/amd_detail/amd_gfx1250_TDM.h>)
  v8i gz8 = (v8i)0;
  __builtin_amdgcn_tensor_load_to_lds(g0, g1, gz, gz, gz8, 0);   // clang-23 form
#else
  __builtin_amdgcn_tensor_load_to_lds(g0, g1, gz, gz, 0);        // ROCm 7.2 form
#endif
}

// ---------------------------------------------------------------------------
// 1) fp32 -> bf16 conversion (8 elems / thread, b128 in & out)
// ---------------------------------------------------------------------------
__global__ void __launch_bounds__(256) cvt_f32_bf16_kernel(
    const float* __restrict__ in, unsigned short* __restrict__ out, long n) {
  long i = ((long)blockIdx.x * blockDim.x + threadIdx.x) * 8;
  if (i >= n) return;
  const float4* p = (const float4*)(in + i);
  float4 a = p[0], c = p[1];
  uint4 o;
  o.x = (unsigned)f2bf(a.x) | ((unsigned)f2bf(a.y) << 16);
  o.y = (unsigned)f2bf(a.z) | ((unsigned)f2bf(a.w) << 16);
  o.z = (unsigned)f2bf(c.x) | ((unsigned)f2bf(c.y) << 16);
  o.w = (unsigned)f2bf(c.z) | ((unsigned)f2bf(c.w) << 16);
  *(uint4*)(out + i) = o;
}

// ---------------------------------------------------------------------------
// 2) bf16 GEMM with TDM double-buffered LDS staging.
//    C[M,2048] = A[M,2048] @ W[2048,2048]^T + bias
//    Block = 256 thr (8 waves), block tile 128x64, wave tile 32x32,
//    K-slab 64, TDM loads A(128x64) + W(64x64) per stage (wave 0 issues).
//    mode 0: bf16 per-head [bh,S,128] | mode 1: bf16 per-head^T [bh,128,S]
//    mode 2: fp32 flat [M,2048]
// ---------------------------------------------------------------------------
__global__ void __launch_bounds__(256) gemm_bf16_kernel(
    const unsigned short* __restrict__ A,
    const unsigned short* __restrict__ W,
    const float* __restrict__ bias,
    unsigned short* __restrict__ out_bf,
    float* __restrict__ out_f,
    int mode) {
  __shared__ alignas(16) unsigned short Abuf[2][128 * KT];  // 2 x 16 KB
  __shared__ alignas(16) unsigned short Wbuf[2][64 * KT];   // 2 x  8 KB

  const int lane = threadIdx.x & 31;
  const int wave = threadIdx.x >> 5;
  const int wm = wave >> 1, wn = wave & 1;
  const int mblk = blockIdx.x * 128;
  const int nblk = blockIdx.y * 64;

  v8f acc[2][2];
#pragma unroll
  for (int i = 0; i < 2; ++i)
#pragma unroll
    for (int j = 0; j < 2; ++j) acc[i][j] = (v8f)0.0f;

  const int nkb = HIDDEN / KT;   // 32 stages
  if (wave == 0) {
    tdm_load_2d((unsigned)(unsigned long long)&Abuf[0][0],
                A + (size_t)mblk * HIDDEN, KT, 128, HIDDEN);
    tdm_load_2d((unsigned)(unsigned long long)&Wbuf[0][0],
                W + (size_t)nblk * HIDDEN, KT, 64, HIDDEN);
  }

  for (int kb = 0; kb < nkb; ++kb) {
    const int cur = kb & 1;
    if (wave == 0) __builtin_amdgcn_s_wait_tensorcnt(0);   // current slab landed
    __syncthreads();
    if (wave == 0 && kb + 1 < nkb) {                       // prefetch next slab
      tdm_load_2d((unsigned)(unsigned long long)&Abuf[cur ^ 1][0],
                  A + (size_t)mblk * HIDDEN + (kb + 1) * KT, KT, 128, HIDDEN);
      tdm_load_2d((unsigned)(unsigned long long)&Wbuf[cur ^ 1][0],
                  W + (size_t)nblk * HIDDEN + (kb + 1) * KT, KT, 64, HIDDEN);
    }
    const unsigned short* As = &Abuf[cur][0];
    const unsigned short* Ws = &Wbuf[cur][0];
#pragma unroll
    for (int ks = 0; ks < KT; ks += 32) {
      v16bf af0 = load_fragA(As + (size_t)(wm * 32)      * KT + ks, KT);
      v16bf af1 = load_fragA(As + (size_t)(wm * 32 + 16) * KT + ks, KT);
      v16bf bf0 = load_fragB(Ws + (size_t)(wn * 32)      * KT + ks, KT);
      v16bf bf1 = load_fragB(Ws + (size_t)(wn * 32 + 16) * KT + ks, KT);
      acc[0][0] = wmma_bf16(af0, bf0, acc[0][0]);
      acc[0][1] = wmma_bf16(af0, bf1, acc[0][1]);
      acc[1][0] = wmma_bf16(af1, bf0, acc[1][0]);
      acc[1][1] = wmma_bf16(af1, bf1, acc[1][1]);
    }
    __syncthreads();   // everyone done with `cur` before it is re-issued
  }

  const int half = lane >> 4, cl = lane & 15;
  const int m0 = mblk + wm * 32, n0 = nblk + wn * 32;
#pragma unroll
  for (int i = 0; i < 2; ++i) {
#pragma unroll
    for (int j = 0; j < 2; ++j) {
#pragma unroll
      for (int r = 0; r < 8; ++r) {
        const int row = m0 + i * 16 + half * 8 + r;
        const int col = n0 + j * 16 + cl;
        const float v = acc[i][j][r] + bias[col];
        if (mode == 2) {
          out_f[(size_t)row * HIDDEN + col] = v;
        } else {
          const int b = row >> 11, s = row & 2047;   // SEQ = 2048
          const int h = col >> 7,  d = col & 127;    // HDIM = 128
          const size_t idx = (mode == 0)
              ? ((size_t)((b * NHEADS + h) * SEQ + s) * HDIM + d)
              : ((size_t)((b * NHEADS + h) * HDIM + d) * SEQ + s);
          out_bf[idx] = f2bf(v);
        }
      }
    }
  }
}

// ---------------------------------------------------------------------------
// 3) Flash attention (causal), bf16 WMMA, fp32 softmax/accum.
// grid = (SEQ/128, B*NHEADS), block = 256 (8 waves); each wave owns 16 query
// rows.  Q [bh,S,128], K [bh,S,128], V transposed [bh,128,S].
// ---------------------------------------------------------------------------
__global__ void __launch_bounds__(256) attn_kernel(
    const unsigned short* __restrict__ Qh,
    const unsigned short* __restrict__ Kh,
    const unsigned short* __restrict__ Vt,
    unsigned short* __restrict__ Ob) {       // bf16 [B, S, HIDDEN]
  __shared__ alignas(16) unsigned short psmem[8][16 * 32];  // per-wave P tile

  const int lane = threadIdx.x & 31;
  const int wave = threadIdx.x >> 5;
  const int half = lane >> 4, cl = lane & 15;
  const int bh = blockIdx.y;
  const int b = bh >> 4, h = bh & 15;
  const int qbase = blockIdx.x * 128 + wave * 16;

  const unsigned short* Qb = Qh + (size_t)bh * SEQ * HDIM;
  const unsigned short* Kb = Kh + (size_t)bh * SEQ * HDIM;
  const unsigned short* Vb = Vt + (size_t)bh * HDIM * SEQ;

  v16bf qf[4];
#pragma unroll
  for (int kk = 0; kk < 4; ++kk)
    qf[kk] = load_fragA(Qb + (size_t)qbase * HDIM + kk * 32, HDIM);

  v8f o[8];
#pragma unroll
  for (int nt = 0; nt < 8; ++nt) o[nt] = (v8f)0.0f;
  float mrow[8], lrow[8];
#pragma unroll
  for (int r = 0; r < 8; ++r) { mrow[r] = -1e30f; lrow[r] = 0.0f; }

  const float scale = 0.08838834764831845f;  // 1/sqrt(128)
  const int nkb = (qbase + 47) >> 5;

  for (int jb = 0; jb < nkb; ++jb) {
    const int jbase = jb * 32;

    v8f s0 = (v8f)0.0f, s1 = (v8f)0.0f;
#pragma unroll
    for (int kk = 0; kk < 4; ++kk) {
      v16bf k0f = load_fragB(Kb + (size_t)jbase        * HDIM + kk * 32, HDIM);
      v16bf k1f = load_fragB(Kb + (size_t)(jbase + 16) * HDIM + kk * 32, HDIM);
      s0 = wmma_bf16(qf[kk], k0f, s0);
      s1 = wmma_bf16(qf[kk], k1f, s1);
    }

    const bool need_mask = (jbase + 31 > qbase);
    float pm0[8], pm1[8];
#pragma unroll
    for (int r = 0; r < 8; ++r) {
      float v0 = s0[r] * scale, v1 = s1[r] * scale;
      const int rowg = qbase + half * 8 + r;
      if (need_mask) {
        if (jbase + cl > rowg)      v0 = -1e30f;
        if (jbase + 16 + cl > rowg) v1 = -1e30f;
      }
      pm0[r] = v0; pm1[r] = v1;
    }

#pragma unroll
    for (int r = 0; r < 8; ++r) {
      float mx = fmaxf(pm0[r], pm1[r]);
      mx = fmaxf(mx, __shfl_xor(mx, 1, 32));
      mx = fmaxf(mx, __shfl_xor(mx, 2, 32));
      mx = fmaxf(mx, __shfl_xor(mx, 4, 32));
      mx = fmaxf(mx, __shfl_xor(mx, 8, 32));
      const float mn    = fmaxf(mrow[r], mx);
      const float alpha = __expf(mrow[r] - mn);
      mrow[r] = mn;
      const float p0 = __expf(pm0[r] - mn);
      const float p1 = __expf(pm1[r] - mn);
      float sum = p0 + p1;
      sum += __shfl_xor(sum, 1, 32);
      sum += __shfl_xor(sum, 2, 32);
      sum += __shfl_xor(sum, 4, 32);
      sum += __shfl_xor(sum, 8, 32);
      lrow[r] = lrow[r] * alpha + sum;
#pragma unroll
      for (int nt = 0; nt < 8; ++nt) o[nt][r] *= alpha;
      const int prow = half * 8 + r;
      psmem[wave][prow * 32 + cl]      = f2bf(p0);
      psmem[wave][prow * 32 + 16 + cl] = f2bf(p1);
    }

    v16bf pf = load_fragA(&psmem[wave][0], 32);
#pragma unroll
    for (int nt = 0; nt < 8; ++nt) {
      v16bf vf = load_fragB(Vb + (size_t)(nt * 16) * SEQ + jbase, SEQ);
      o[nt] = wmma_bf16(pf, vf, o[nt]);
    }
  }

#pragma unroll
  for (int r = 0; r < 8; ++r) {
    const float inv = lrow[r] > 0.0f ? 1.0f / lrow[r] : 0.0f;
    const int srow = qbase + half * 8 + r;
#pragma unroll
    for (int nt = 0; nt < 8; ++nt) {
      const float v = o[nt][r] * inv;
      const size_t idx =
          (size_t)(b * SEQ + srow) * HIDDEN + h * HDIM + nt * 16 + cl;
      Ob[idx] = f2bf(v);
    }
  }
}

// ---------------------------------------------------------------------------
// launch
// ---------------------------------------------------------------------------
extern "C" void kernel_launch(void* const* d_in, const int* in_sizes, int n_in,
                              void* d_out, int out_size, void* d_ws, size_t ws_size,
                              hipStream_t stream) {
  const float* X  = (const float*)d_in[0];
  const float* wq = (const float*)d_in[1];
  const float* bq = (const float*)d_in[2];
  const float* wk = (const float*)d_in[3];
  const float* bk = (const float*)d_in[4];
  const float* wv = (const float*)d_in[5];
  const float* bv = (const float*)d_in[6];
  const float* wo = (const float*)d_in[7];
  const float* bo = (const float*)d_in[8];

  const size_t nX = (size_t)MROWS * HIDDEN;    // 8,388,608
  const size_t nW = (size_t)HIDDEN * HIDDEN;   // 4,194,304

  unsigned short* Xb  = (unsigned short*)d_ws;
  unsigned short* Wqb = Xb  + nX;
  unsigned short* Wkb = Wqb + nW;
  unsigned short* Wvb = Wkb + nW;
  unsigned short* Wob = Wvb + nW;
  unsigned short* Qh  = Wob + nW;
  unsigned short* Kh  = Qh  + nX;
  unsigned short* VtB = Kh  + nX;
  unsigned short* Ab  = VtB + nX;   // total ~112 MB of bf16 scratch

  cvt_f32_bf16_kernel<<<(unsigned)(nX / 8 / 256), 256, 0, stream>>>(X,  Xb,  (long)nX);
  cvt_f32_bf16_kernel<<<(unsigned)(nW / 8 / 256), 256, 0, stream>>>(wq, Wqb, (long)nW);
  cvt_f32_bf16_kernel<<<(unsigned)(nW / 8 / 256), 256, 0, stream>>>(wk, Wkb, (long)nW);
  cvt_f32_bf16_kernel<<<(unsigned)(nW / 8 / 256), 256, 0, stream>>>(wv, Wvb, (long)nW);
  cvt_f32_bf16_kernel<<<(unsigned)(nW / 8 / 256), 256, 0, stream>>>(wo, Wob, (long)nW);

  dim3 gg(MROWS / 128, HIDDEN / 64);   // 32 x 32
  gemm_bf16_kernel<<<gg, 256, 0, stream>>>(Xb, Wqb, bq, Qh,  nullptr, 0);
  gemm_bf16_kernel<<<gg, 256, 0, stream>>>(Xb, Wkb, bk, Kh,  nullptr, 0);
  gemm_bf16_kernel<<<gg, 256, 0, stream>>>(Xb, Wvb, bv, VtB, nullptr, 1);

  attn_kernel<<<dim3(SEQ / 128, BATCH * NHEADS), 256, 0, stream>>>(Qh, Kh, VtB, Ab);

  gemm_bf16_kernel<<<gg, 256, 0, stream>>>(Ab, Wob, bo, nullptr, (float*)d_out, 2);
}
